// LocalAggregator_52836687675884
// MI455X (gfx1250) — matrix operand order repaired and checked by reading, not compiled, not run on hardware
//
#include <hip/hip_runtime.h>
#include <cstdint>
#include <cstddef>

#define BB 32
#define NN 128
#define DD 256
#define TT 64
#define HS  264     // bf16 row stride for sh_h / sh_ha: 528B = 33*16 (b128-aligned), 132 dw = 4 mod 64 banks
#define HTS 136     // bf16 row stride for sh_ht / al_bf: 272B = 17*16, 68 dw = 4 mod 64 banks
#define LRELU 0.2f
#define MASKVAL (-9e15f)

typedef __attribute__((ext_vector_type(16))) __bf16 v16bf;
typedef __attribute__((ext_vector_type(8)))  float  v8f;

union Frag { uint4 q[2]; v16bf v; };   // 32B: two ds_load_b128 -> one WMMA operand

// ---- shared memory layout (bytes, every region 16B aligned) ----
#define OFF_H   0                         // bf16 h[b]      : 128 x 264      = 67584
#define OFF_HT  67584                     // bf16 h[b]^T    : 256 x 136      = 69632
#define OFF_HA  (OFF_HT + 69632)          // bf16 A_r mats  : 5 x 16 x 264   = 42240
#define OFF_E2  (OFF_HA + 42240)          // f32 e2[16][128][5]              = 40960
#define OFF_ALF (OFF_E2 + 40960)          // f32 logits alpha[16][128]       = 8192
#define OFF_ALB (OFF_ALF + 8192)          // bf16 softmax(alpha): 16 x 136   = 4352
#define OFF_AP  (OFF_ALB + 4352)          // f32 a_params[256][5]            = 5120
#define OFF_F   (OFF_AP + 5120)           // f32 te_freq[64]                 = 256
#define OFF_P   (OFF_F + 256)             // f32 te_phase[64]                = 256
#define OFF_IW  (OFF_P + 256)             // f32 iw[64][5]                   = 1280
#define SMEM_TOTAL (OFF_IW + 1280)        // 239872 bytes dynamic LDS

static __device__ __forceinline__ uint16_t f2bf(float f) {
    uint32_t u = __builtin_bit_cast(uint32_t, f);
    u += 0x7FFFu + ((u >> 16) & 1u);          // round-to-nearest-even
    return (uint16_t)(u >> 16);
}
static __device__ __forceinline__ float bf2f(uint16_t b) {
    return __builtin_bit_cast(float, (uint32_t)b << 16);
}
static __device__ __forceinline__ __bf16 bfbits(uint16_t b) {
    return __builtin_bit_cast(__bf16, b);
}
static __device__ __forceinline__ v8f wmma_bf16(const Frag& a, const Frag& b, v8f c) {
    return __builtin_amdgcn_wmma_f32_16x16x32_bf16(false, a.v, false, b.v,
                                                   (short)0, c, false, false);
}

__global__ void __launch_bounds__(128)
localagg_fused_kernel(const float* __restrict__ hidden,
                      const float* __restrict__ A_interval,
                      const float* __restrict__ a_params,
                      const float* __restrict__ iw_params,
                      const float* __restrict__ te_freq,
                      const float* __restrict__ te_phase,
                      const int*   __restrict__ adj,
                      float* __restrict__ out)
{
    extern __shared__ char smem[];
    uint16_t* sh_h  = (uint16_t*)(smem + OFF_H);
    uint16_t* sh_ht = (uint16_t*)(smem + OFF_HT);
    uint16_t* sh_ha = (uint16_t*)(smem + OFF_HA);
    float*    sh_e2 = (float*)(smem + OFF_E2);
    float*    sh_al = (float*)(smem + OFF_ALF);
    uint16_t* al_bf = (uint16_t*)(smem + OFF_ALB);
    float*    sh_a  = (float*)(smem + OFF_AP);
    float*    sh_f  = (float*)(smem + OFF_F);
    float*    sh_p  = (float*)(smem + OFF_P);
    float*    sh_iw = (float*)(smem + OFF_IW);

    const int tid  = threadIdx.x;
    const int lane = tid & 31;
    const int wave = tid >> 5;
    const int lhi  = lane >> 4;      // 0/1 half of wave
    const int llo  = lane & 15;
    const int b    = blockIdx.y;
    const int i00  = blockIdx.x * 16;

    // ---------- Phase 1: small params + logit-mask init ----------
    for (int k = tid; k < DD * 5; k += 128) sh_a[k] = a_params[k];
    for (int k = tid; k < TT;     k += 128) { sh_f[k] = te_freq[k]; sh_p[k] = te_phase[k]; }
    for (int k = tid; k < TT * 5; k += 128) sh_iw[k] = iw_params[k];
    for (int k = tid; k < 16 * NN; k += 128) sh_al[k] = MASKVAL;
    __syncthreads();

    // ---------- Phase 2a: stage h[b] -> bf16 row-major + transposed ----------
    const float4* hsrc = (const float4*)(hidden + (size_t)b * NN * DD);
    for (int k = tid; k < (NN * DD) / 4; k += 128) {
        float4 v = hsrc[k];
        const int row = (k * 4) / DD, col = (k * 4) % DD;
        uint16_t b0 = f2bf(v.x), b1 = f2bf(v.y), b2 = f2bf(v.z), b3 = f2bf(v.w);
        uint32_t* dst = (uint32_t*)(sh_h + row * HS + col);
        dst[0] = (uint32_t)b0 | ((uint32_t)b1 << 16);
        dst[1] = (uint32_t)b2 | ((uint32_t)b3 << 16);
        sh_ht[(col + 0) * HTS + row] = b0;
        sh_ht[(col + 1) * HTS + row] = b1;
        sh_ht[(col + 2) * HTS + row] = b2;
        sh_ht[(col + 3) * HTS + row] = b3;
    }
    // ---------- Phase 2b: A_r[i,d] = h[i00+i,d] * a[d,r] -> bf16 LDS ----------
    for (int k = tid; k < 5 * 16 * (DD / 4); k += 128) {
        const int r  = k / (16 * (DD / 4));
        const int rm = k % (16 * (DD / 4));
        const int il = rm / (DD / 4);
        const int dq = (rm % (DD / 4)) * 4;
        float4 v = ((const float4*)(hidden + ((size_t)b * NN + i00 + il) * DD))[dq / 4];
        uint16_t b0 = f2bf(v.x * sh_a[(dq + 0) * 5 + r]);
        uint16_t b1 = f2bf(v.y * sh_a[(dq + 1) * 5 + r]);
        uint16_t b2 = f2bf(v.z * sh_a[(dq + 2) * 5 + r]);
        uint16_t b3 = f2bf(v.w * sh_a[(dq + 3) * 5 + r]);
        uint32_t* dst = (uint32_t*)(sh_ha + (r * 16 + il) * HS + dq);
        dst[0] = (uint32_t)b0 | ((uint32_t)b1 << 16);
        dst[1] = (uint32_t)b2 | ((uint32_t)b3 << 16);
    }
    // ---------- Phase 2c: e2[i,j,r] = sum_t cos(x*f_t + p_t) * iw[t,r] ----------
    const float* Aib = A_interval + ((size_t)b * NN + i00) * NN;
    for (int pr = tid; pr < 16 * NN; pr += 128) {
        const int il = pr >> 7, j = pr & 127;
        const float x = Aib[il * NN + j];
        float a0 = 0.f, a1 = 0.f, a2 = 0.f, a3 = 0.f, a4 = 0.f;
        #pragma unroll 8
        for (int t = 0; t < TT; ++t) {
            float c = __cosf(fmaf(x, sh_f[t], sh_p[t]));
            a0 = fmaf(c, sh_iw[t * 5 + 0], a0);
            a1 = fmaf(c, sh_iw[t * 5 + 1], a1);
            a2 = fmaf(c, sh_iw[t * 5 + 2], a2);
            a3 = fmaf(c, sh_iw[t * 5 + 3], a3);
            a4 = fmaf(c, sh_iw[t * 5 + 4], a4);
        }
        float* dst = sh_e2 + pr * 5;
        dst[0] = a0; dst[1] = a1; dst[2] = a2; dst[3] = a3; dst[4] = a4;
    }
    __syncthreads();

    // ---------- Phase D: logit GEMMs, B-frag reused across 5 branches ----------
    const int* adjb = adj + ((size_t)b * NN + i00) * NN;
    for (int t = 0; t < 2; ++t) {
        const int jt = wave * 2 + t;              // 8 j-tiles over 4 waves
        v8f acc[5] = {};
        const uint16_t* brow = sh_h + (jt * 16 + llo) * HS + lhi * 8;
        for (int kc = 0; kc < 8; ++kc) {          // K = 256 in chunks of 32
            Frag fb;
            const uint16_t* bp = brow + kc * 32;  // two 8-bf16 runs per lane, 16 apart
            fb.q[0] = *(const uint4*)bp;
            fb.q[1] = *(const uint4*)(bp + 16);
            #pragma unroll
            for (int r = 0; r < 5; ++r) {
                const uint16_t* ap = sh_ha + (r * 16 + llo) * HS + kc * 32 + lhi * 8;
                Frag fa;
                fa.q[0] = *(const uint4*)ap;          // K offsets 0..7
                fa.q[1] = *(const uint4*)(ap + 16);   // K offsets 16..23
                acc[r] = wmma_bf16(fa, fb, acc[r]);
            }
        }
        const int j = jt * 16 + llo;
        int adjv[8];
        #pragma unroll
        for (int p = 0; p < 8; ++p) adjv[p] = adjb[(p + lhi * 8) * NN + j];
        #pragma unroll
        for (int r = 0; r < 5; ++r) {
            #pragma unroll
            for (int p = 0; p < 8; ++p) {
                if (adjv[p] == r + 1) {
                    const int il = p + lhi * 8;
                    float e = acc[r][p] + sh_e2[(il * NN + j) * 5 + r];
                    e = (e >= 0.f) ? e : LRELU * e;
                    sh_al[il * NN + j] = e;        // unique writer per (i,j)
                }
            }
        }
    }
    __syncthreads();

    // ---------- Phase E: row softmax over j; emit bf16 alpha ----------
    {
        const int il = tid >> 3, seg = tid & 7;
        const float* row = sh_al + il * NN;
        float m = -__builtin_inff();
        #pragma unroll
        for (int k = 0; k < 16; ++k) m = fmaxf(m, row[seg * 16 + k]);
        m = fmaxf(m, __shfl_xor(m, 1, 8));
        m = fmaxf(m, __shfl_xor(m, 2, 8));
        m = fmaxf(m, __shfl_xor(m, 4, 8));
        float loc[16], s = 0.f;
        #pragma unroll
        for (int k = 0; k < 16; ++k) { loc[k] = __expf(row[seg * 16 + k] - m); s += loc[k]; }
        s += __shfl_xor(s, 1, 8);
        s += __shfl_xor(s, 2, 8);
        s += __shfl_xor(s, 4, 8);
        const float inv = 1.f / s;
        #pragma unroll
        for (int k = 0; k < 16; ++k)
            al_bf[il * HTS + seg * 16 + k] = f2bf(loc[k] * inv);
    }
    __syncthreads();

    // ---------- Phase F: out[i,d] = alpha @ h  (A frags hoisted) ----------
    Frag fA[4];
    #pragma unroll
    for (int kc = 0; kc < 4; ++kc) {              // K = 128 (j) in chunks of 32
        const uint16_t* ap = al_bf + llo * HTS + kc * 32 + lhi * 8;
        fA[kc].q[0] = *(const uint4*)ap;
        fA[kc].q[1] = *(const uint4*)(ap + 16);
    }
    for (int t = 0; t < 4; ++t) {
        const int dt = wave * 4 + t;              // 16 d-tiles over 4 waves
        v8f acc = {};
        #pragma unroll
        for (int kc = 0; kc < 4; ++kc) {
            const uint16_t* bp = sh_ht + (dt * 16 + llo) * HTS + kc * 32 + lhi * 8;
            Frag fb;
            fb.q[0] = *(const uint4*)bp;
            fb.q[1] = *(const uint4*)(bp + 16);
            acc = wmma_bf16(fA[kc], fb, acc);
        }
        #pragma unroll
        for (int p = 0; p < 8; ++p) {
            const int il = p + lhi * 8;
            const int d  = dt * 16 + llo;
            out[((size_t)b * NN + (i00 + il)) * DD + d] = acc[p];
        }
    }
}

extern "C" void kernel_launch(void* const* d_in, const int* in_sizes, int n_in,
                              void* d_out, int out_size, void* d_ws, size_t ws_size,
                              hipStream_t stream) {
    (void)in_sizes; (void)n_in; (void)out_size; (void)d_ws; (void)ws_size;
    const float* hidden     = (const float*)d_in[0];
    const float* A_interval = (const float*)d_in[1];
    const float* a_params   = (const float*)d_in[2];
    const float* iw_params  = (const float*)d_in[3];
    const float* te_freq    = (const float*)d_in[4];
    const float* te_phase   = (const float*)d_in[5];
    const int*   adj        = (const int*)d_in[6];
    float* out = (float*)d_out;

    dim3 grid(NN / 16, BB);   // 8 i-tiles x 32 batches
    dim3 block(128);          // 4 wave32s
    localagg_fused_kernel<<<grid, block, SMEM_TOTAL, stream>>>(
        hidden, A_interval, a_params, iw_params, te_freq, te_phase, adj, out);
}
